// DeepHedgingAgent_44890998178148
// MI455X (gfx1250) — matrix operand order repaired and checked by reading, not compiled, not run on hardware
//
#include <hip/hip_runtime.h>

// DeepHedgingAgent: 2-layer GRU (H=32) + MLP head, B=8192, T=2048.
// One wave (wave32) owns a 16-sample batch tile and iterates the full
// recurrence with all weights resident in VGPRs as WMMA A-operands.
// All matmuls computed transposed: gates^T (gate x batch) = W (gate x K) @ x^T (K x batch),
// so C/D tiles (rows across VGPRs, batch across lanes) are elementwise-aligned
// with the f32 hidden-state tiles.

typedef __attribute__((ext_vector_type(16))) _Float16 v16h;
typedef __attribute__((ext_vector_type(8)))  float    v8f;

#define T_STEPS 2048

static __device__ __forceinline__ v8f wmma16(v16h a, v16h b, v8f c) {
  // D = A(16x32 f16) * B(32x16 f16) + C(16x16 f32)
  return __builtin_amdgcn_wmma_f32_16x16x32_f16(false, a, false, b, (short)0, c, false, false);
}

static __device__ __forceinline__ v8f zero8() {
  v8f c;
#pragma unroll
  for (int i = 0; i < 8; ++i) c[i] = 0.0f;
  return c;
}
static __device__ __forceinline__ v16h zero16h() {
  v16h a;
#pragma unroll
  for (int i = 0; i < 16; ++i) a[i] = (_Float16)0.0f;
  return a;
}

// exchange with lane ^ 16 (wave32)
static __device__ __forceinline__ float xswap16(float x) { return __shfl_xor(x, 16, 32); }

// gfx1250 has v_tanh_f32 (TRANS32) — confirmed in round-2 disassembly.
static __device__ __forceinline__ float tanh_fast(float x) {
#if __has_builtin(__builtin_amdgcn_tanhf)
  return __builtin_amdgcn_tanhf(x);
#else
  return 1.0f - 2.0f * __builtin_amdgcn_rcpf(__expf(2.0f * x) + 1.0f);
#endif
}
// sigmoid(x) = 0.5*tanh(x/2) + 0.5 : ONE transcendental (v_tanh) instead of
// exp + rcp, and a 3-op dependency chain (mul -> tanh -> fma).
static __device__ __forceinline__ float sigm(float x) {
  return __builtin_fmaf(0.5f, tanh_fast(0.5f * x), 0.5f);
}

// 16-bit A-operand tile (16 rows x K=32), this code's contiguous k order (matches B):
// lane l: row = rowBase + (l&15); element e -> K = e + 16*(l>>4).
// Columns k >= KD are zero-padded; optionally one pad column carries a bias.
static __device__ __forceinline__ v16h load_a_tile(const float* __restrict__ W, int KD,
                                                   int row, int g, float biasVal, int biasSlot) {
  v16h a;
#pragma unroll
  for (int e = 0; e < 16; ++e) {
    int k = e + 16 * g;
    float v;
    if (k < KD)             v = W[row * KD + k];
    else if (k == biasSlot) v = biasVal;
    else                    v = 0.0f;
    a[e] = (_Float16)v;
  }
  return a;
}

// f32 C-layout bias tile: element e (lane group g) holds bias[rowBase + e + 8g].
static __device__ __forceinline__ v8f load_bias_tile(const float* __restrict__ b0,
                                                     const float* __restrict__ b1,
                                                     int rowBase, int g) {
  v8f c;
#pragma unroll
  for (int e = 0; e < 8; ++e) {
    int r = rowBase + e + 8 * g;
    float v = b0[r];
    if (b1) v += b1[r];
    c[e] = v;
  }
  return c;
}

// Build a 32x16 f16 B-operand (K = hidden unit, N = batch) from the two f32
// C-layout tiles of the state (t0: rows 0..15, t1: rows 16..31).
// B layout: lanes 0-15 hold K=0..15, lanes 16-31 hold K=16..31, element e -> K = 16g+e.
// The half-wave mismatch (rows 8..15 / 16..23 live in the partner lane group)
// is resolved with a single lane^16 exchange per register.
static __device__ __forceinline__ v16h make_b_state(const v8f& t0, const v8f& t1, bool hi) {
  v16h b;
#pragma unroll
  for (int e = 0; e < 8; ++e) {
    float y  = hi ? t0[e] : t1[e];  // send partner what it needs
    float sh = xswap16(y);          // lo-group receives partner t0[e] (rows 8..15)
                                    // hi-group receives partner t1[e] (rows 16..23)
    float lo = hi ? sh    : t0[e];  // K = 16g + e
    float h2 = hi ? t1[e] : sh;     // K = 16g + 8 + e
    b[e]     = (_Float16)lo;
    b[e + 8] = (_Float16)h2;
  }
  return b;
}

// hid (16 units) as B-operand: K=0..15 = hid rows, K=16 carries 1.0 (bias lane for b2).
static __device__ __forceinline__ v16h make_b_hid(const v8f& t0, bool hi) {
  v16h b;
#pragma unroll
  for (int e = 0; e < 8; ++e) {
    float sh = xswap16(t0[e]);                       // lo-group gets rows 8..15
    float lo = hi ? ((e == 0) ? 1.0f : 0.0f) : t0[e];
    float h2 = hi ? 0.0f : sh;
    b[e]     = (_Float16)lo;
    b[e + 8] = (_Float16)h2;
  }
  return b;
}

// feature vector [s, t, d, 1(bias lane), 0...] as B-operand (K padded to 32)
static __device__ __forceinline__ v16h make_b_feats(float s, float tval, float d, bool hi) {
  v16h b = zero16h();
  if (!hi) {
    b[0] = (_Float16)s;
    b[1] = (_Float16)tval;
    b[2] = (_Float16)d;
    b[3] = (_Float16)1.0f;  // multiplies the folded biases stored at K=3 of Wih0 tiles
  }
  return b;
}

__global__ __launch_bounds__(32, 1) void deep_hedge_gru(
    const float* __restrict__ paths,
    const float* __restrict__ Wih0, const float* __restrict__ Whh0,
    const float* __restrict__ bih0, const float* __restrict__ bhh0,
    const float* __restrict__ Wih1, const float* __restrict__ Whh1,
    const float* __restrict__ bih1, const float* __restrict__ bhh1,
    const float* __restrict__ W1, const float* __restrict__ b1,
    const float* __restrict__ W2, const float* __restrict__ b2,
    float* __restrict__ out)
{
  const int lane = threadIdx.x & 31;
  const bool hi  = lane >= 16;
  const int  g   = hi ? 1 : 0;
  const int  m   = lane & 15;
  const int  batchBase = blockIdx.x * 16;

  // ---- weights as resident f16 WMMA A-operands ----
  v16h a_wi0[6], a_wh0[6], a_wi1[6], a_wh1[6];
#pragma unroll
  for (int tt = 0; tt < 6; ++tt) {
    const int row = tt * 16 + m;
    // gate rows: [0,32)=r  [32,64)=z  [64,96)=n
    // r/z: fold (bih0+bhh0) into pad column K=3; n: fold bih0 only (bhh0_n must stay separate).
    const float bfold = (tt < 4) ? (bih0[row] + bhh0[row]) : bih0[row];
    a_wi0[tt] = load_a_tile(Wih0, 3,  row, g, bfold, 3);
    a_wh0[tt] = load_a_tile(Whh0, 32, row, g, 0.0f, -1);
    a_wi1[tt] = load_a_tile(Wih1, 32, row, g, 0.0f, -1);
    a_wh1[tt] = load_a_tile(Whh1, 32, row, g, 0.0f, -1);
  }
  const v16h a_w1 = load_a_tile(W1, 32, m, g, 0.0f, -1);

  // W2 duplicated into output rows 0 AND 8 so both half-wave groups receive
  // delta[n] in accumulator element 0 (no shuffle on the feedback path).
  // b2 folded at K=16 (paired with the 1.0 lane of make_b_hid).
  v16h a_w2;
  {
    const bool act = (m == 0) || (m == 8);
#pragma unroll
    for (int e = 0; e < 16; ++e) {
      int k = e + 16 * g;
      float v = 0.0f;
      if (act) {
        if (k < 16)       v = W2[k];
        else if (k == 16) v = b2[0];
      }
      a_w2[e] = (_Float16)v;
    }
  }

  // ---- bias C-init tiles ----
  v8f c_nh0[2], c_rz1[4], c_ni1[2], c_nh1[2];
#pragma unroll
  for (int hb = 0; hb < 2; ++hb) {
    c_nh0[hb] = load_bias_tile(bhh0, nullptr, 64 + 16 * hb, g);
    c_ni1[hb] = load_bias_tile(bih1, nullptr, 64 + 16 * hb, g);
    c_nh1[hb] = load_bias_tile(bhh1, nullptr, 64 + 16 * hb, g);
  }
#pragma unroll
  for (int tt = 0; tt < 4; ++tt) c_rz1[tt] = load_bias_tile(bih1, bhh1, 16 * tt, g);
  const v8f c_b1 = load_bias_tile(b1, nullptr, 0, g);

  // ---- recurrent state ----
  v8f  h0t[2] = {zero8(), zero8()};
  v8f  h1t[2] = {zero8(), zero8()};
  v16h b_h0 = zero16h();   // h0 as B-operand (reused as next step's recurrent input)
  v16h b_h1 = zero16h();
  float d_prev = 0.0f;

  const float* __restrict__ srow = paths + (size_t)(batchBase + m) * T_STEPS;
  float* __restrict__       orow = out   + (size_t)(batchBase + m) * T_STEPS;
  const float tscale = 1.0f / (float)(T_STEPS - 1);

  float s_cur = srow[0];

#pragma unroll 1
  for (int t = 0; t < T_STEPS; ++t) {
    // branch-free prefetch of next sample (wraps harmlessly on the last step)
    const float s_next = srow[(t + 1) & (T_STEPS - 1)];
    const v16h b_x = make_b_feats(s_cur, (float)t * tscale, d_prev, hi);

    v8f acc_r[2], acc_z[2], acc_ni[2], acc_nh[2];

    // ---- GRU layer 0 (12 WMMAs) ----
#pragma unroll
    for (int hb = 0; hb < 2; ++hb) {
      acc_r[hb]  = wmma16(a_wh0[hb],     b_h0, wmma16(a_wi0[hb],     b_x, zero8()));
      acc_z[hb]  = wmma16(a_wh0[2 + hb], b_h0, wmma16(a_wi0[2 + hb], b_x, zero8()));
      acc_ni[hb] = wmma16(a_wi0[4 + hb], b_x,  zero8());
      acc_nh[hb] = wmma16(a_wh0[4 + hb], b_h0, c_nh0[hb]);
    }
#pragma unroll
    for (int hb = 0; hb < 2; ++hb)
#pragma unroll
      for (int e = 0; e < 8; ++e) {
        float r  = sigm(acc_r[hb][e]);
        float z  = sigm(acc_z[hb][e]);
        float nn = tanh_fast(acc_ni[hb][e] + r * acc_nh[hb][e]);
        h0t[hb][e] = nn + z * (h0t[hb][e] - nn);   // (1-z)*n + z*h
      }
    b_h0 = make_b_state(h0t[0], h0t[1], hi);

    // ---- GRU layer 1 (12 WMMAs) ----
#pragma unroll
    for (int hb = 0; hb < 2; ++hb) {
      acc_r[hb]  = wmma16(a_wh1[hb],     b_h1, wmma16(a_wi1[hb],     b_h0, c_rz1[hb]));
      acc_z[hb]  = wmma16(a_wh1[2 + hb], b_h1, wmma16(a_wi1[2 + hb], b_h0, c_rz1[2 + hb]));
      acc_ni[hb] = wmma16(a_wi1[4 + hb], b_h0, c_ni1[hb]);
      acc_nh[hb] = wmma16(a_wh1[4 + hb], b_h1, c_nh1[hb]);
    }
#pragma unroll
    for (int hb = 0; hb < 2; ++hb)
#pragma unroll
      for (int e = 0; e < 8; ++e) {
        float r  = sigm(acc_r[hb][e]);
        float z  = sigm(acc_z[hb][e]);
        float nn = tanh_fast(acc_ni[hb][e] + r * acc_nh[hb][e]);
        h1t[hb][e] = nn + z * (h1t[hb][e] - nn);
      }
    b_h1 = make_b_state(h1t[0], h1t[1], hi);

    // ---- MLP head (2 WMMAs) ----
    v8f hid = wmma16(a_w1, b_h1, c_b1);
#pragma unroll
    for (int e = 0; e < 8; ++e) hid[e] = fmaxf(hid[e], 0.0f);
    const v16h b_hd = make_b_hid(hid, hi);
    const v8f  dacc = wmma16(a_w2, b_hd, zero8());
    const float delta = tanh_fast(dacc[0]);  // valid in BOTH lane groups (rows 0 & 8 dup'd)
    d_prev = delta;

    if (!hi) orow[t] = delta;
    s_cur = s_next;
  }
}

extern "C" void kernel_launch(void* const* d_in, const int* in_sizes, int n_in,
                              void* d_out, int out_size, void* d_ws, size_t ws_size,
                              hipStream_t stream) {
  const float* paths = (const float*)d_in[0];
  const float* Wih0  = (const float*)d_in[1];
  const float* Whh0  = (const float*)d_in[2];
  const float* bih0  = (const float*)d_in[3];
  const float* bhh0  = (const float*)d_in[4];
  const float* Wih1  = (const float*)d_in[5];
  const float* Whh1  = (const float*)d_in[6];
  const float* bih1  = (const float*)d_in[7];
  const float* bhh1  = (const float*)d_in[8];
  const float* W1    = (const float*)d_in[9];
  const float* b1    = (const float*)d_in[10];
  const float* W2    = (const float*)d_in[11];
  const float* b2    = (const float*)d_in[12];
  float* out = (float*)d_out;

  const int B = in_sizes[0] / T_STEPS;   // 8192
  const int nblocks = B / 16;            // 512 independent waves
  deep_hedge_gru<<<nblocks, 32, 0, stream>>>(paths, Wih0, Whh0, bih0, bhh0,
                                             Wih1, Whh1, bih1, bhh1,
                                             W1, b1, W2, b2, out);
}
